// Vim_Block_35467839930877
// MI455X (gfx1250) — compile-verified, matrix operand order checked by reading, not run on hardware
//
#include <hip/hip_runtime.h>
#include <hip/hip_bf16.h>
#include <math.h>

typedef __attribute__((ext_vector_type(16))) _Float16 v16h;
typedef __attribute__((ext_vector_type(8)))  _Float16 v8h;
typedef __attribute__((ext_vector_type(8)))  float    v8f;

#define B_    8
#define C_    256
#define L_    1024
#define DIN   512
#define DST   16
#define NTOK  (B_ * L_)      // 8192 tokens
#define NTTILE (NTOK / 16)   // 512 tiles of 16 tokens
#define NCH   8              // scan chunks
#define CL    (L_ / NCH)     // 128 per chunk

// ---------------- workspace layout (bytes) ----------------
#define OFF_WINT   0x0000000ull   // 1024*256 f16  (512KB)
#define OFF_WXT    0x0080000ull   // 48*512 f16
#define OFF_WDTT   0x0090000ull   // 512*32 f16 (K padded 16->32 with zeros)
#define OFF_WOUTT  0x00A0000ull   // 256*512 f16
#define OFF_XSPRE  0x0100000ull   // 8192*512 f32 (16MB)
#define OFF_Z      0x1100000ull
#define OFF_XS     0x2100000ull
#define OFF_DT     0x3100000ull
#define OFF_YS     0x4100000ull
#define OFF_BM     0x5100000ull   // 8192*16 f32
#define OFF_CM     0x5180000ull
#define OFF_APROD  0x5200000ull   // 8*8*512*16 f32 (2MB)
#define OFF_HEND   0x5400000ull
#define OFF_HIN    0x5600000ull
#define OFF_XSF16  0x5800000ull   // 8192*512 f16 (8MB)

// ---------------- WMMA fragment helpers (ISA 7.12.2 layouts) ----------------
__device__ __forceinline__ v16h frag16(const _Float16* lo, const _Float16* hi) {
  v8h a = *(const v8h*)lo;
  v8h b = *(const v8h*)hi;
  v16h r;
#pragma unroll
  for (int i = 0; i < 8; ++i) { r[i] = a[i]; r[i + 8] = b[i]; }
  return r;
}
// A: 16xK tile, row-major [16][stride] halves in LDS. lane&15 = M row.
__device__ __forceinline__ v16h loadA(const _Float16* tile, int stride, int k0, int lane) {
  const _Float16* p = tile + (lane & 15) * stride + k0 + ((lane & 16) ? 8 : 0);
  return frag16(p, p + 16);
}
// B: from pre-transposed weight WT[n][stride] halves. lane&15 = N col.
__device__ __forceinline__ v16h loadB(const _Float16* WT, int stride, int n0, int k0, int lane) {
  const _Float16* p = WT + (size_t)(n0 + (lane & 15)) * stride + k0 + ((lane & 16) ? 8 : 0);
  return frag16(p, p + 16);
}
#define WMMA(a, b, c) __builtin_amdgcn_wmma_f32_16x16x32_f16(false, (a), false, (b), (short)0, (c), false, false)

__device__ __forceinline__ float sigmoidf_(float v) { return 1.0f / (1.0f + __expf(-v)); }
__device__ __forceinline__ float softplusf_(float v) { return (v > 20.f) ? v : log1pf(__expf(v)); }

// gfx1250 async DMA: copy 16B from global to LDS, tracked by ASYNCcnt
__device__ __forceinline__ void async_g2l_b128(unsigned lds_dst, unsigned long long gaddr) {
  asm volatile("global_load_async_to_lds_b128 %0, %1, off"
               :: "v"(lds_dst), "v"(gaddr) : "memory");
}
__device__ __forceinline__ void wait_asynccnt0() {
  asm volatile("s_wait_asynccnt 0x0" ::: "memory");
}

// ---------------- 0: transpose + f16-convert all GEMM weights ----------------
__global__ __launch_bounds__(256) void k_prep(const float* Win, const float* Wx,
                                              const float* Wdt, const float* Wout,
                                              _Float16* WinT, _Float16* WxT,
                                              _Float16* WdtT, _Float16* WoutT) {
  int t = blockIdx.x * 256 + threadIdx.x;
  int stride = gridDim.x * 256;
  for (int i = t; i < 1024 * 256; i += stride) {           // WinT[n][k], k<256
    int n = i >> 8, k = i & 255;
    WinT[i] = (_Float16)Win[k * 1024 + n];
  }
  for (int i = t; i < 48 * 512; i += stride) {             // WxT[n][k], k<512
    int n = i >> 9, k = i & 511;
    WxT[i] = (_Float16)Wx[k * 48 + n];
  }
  for (int i = t; i < 512 * 32; i += stride) {             // WdtT[n][k], K padded to 32
    int n = i >> 5, k = i & 31;
    WdtT[i] = (k < 16) ? (_Float16)Wdt[k * 512 + n] : (_Float16)0.0f;
  }
  for (int i = t; i < 256 * 512; i += stride) {            // WoutT[n][k], k<512
    int n = i >> 9, k = i & 511;
    WoutT[i] = (_Float16)Wout[k * 256 + n];
  }
}

// ---------------- 1: LayerNorm + in-projection GEMM (256 -> 1024) ----------------
__global__ __launch_bounds__(256) void k_ln_inproj(const float* x, const float* lnw, const float* lnb,
                                                   const _Float16* WinT, float* xs_pre, float* z) {
  __shared__ __align__(16) float u[16 * 257];
  __shared__ __align__(16) _Float16 hf[16 * 264];
  __shared__ float psum[16][17], psq[16][17];
  __shared__ float mu[16], rs[16];
  int tid = threadIdx.x, lane = tid & 31, wave = tid >> 5;
  int b = blockIdx.x >> 6, l0 = (blockIdx.x & 63) * 16;

  // stage u tile: u[j][c] = x[b, c, l0+j]
  for (int i = tid; i < 16 * 256; i += 256) {
    int c = i >> 4, j = i & 15;
    u[j * 257 + c] = x[(size_t)b * C_ * L_ + (size_t)c * L_ + l0 + j];
  }
  __syncthreads();
  { // parallel mean/var: 16 partials per token, then 16-way reduce
    int j = tid >> 4, p = tid & 15;
    float s = 0.f, q = 0.f;
#pragma unroll
    for (int c = p * 16; c < p * 16 + 16; ++c) { float v = u[j * 257 + c]; s += v; q += v * v; }
    psum[j][p] = s; psq[j][p] = q;
  }
  __syncthreads();
  if (tid < 16) {
    float s = 0.f, q = 0.f;
#pragma unroll
    for (int p = 0; p < 16; ++p) { s += psum[tid][p]; q += psq[tid][p]; }
    float m = s * (1.0f / 256.0f);
    mu[tid] = m;
    rs[tid] = rsqrtf(q * (1.0f / 256.0f) - m * m + 1e-5f);
  }
  __syncthreads();
  for (int i = tid; i < 16 * 256; i += 256) {
    int j = i >> 8, c = i & 255;
    float h = (u[j * 257 + c] - mu[j]) * rs[j] * lnw[c] + lnb[c];
    hf[j * 264 + c] = (_Float16)h;
  }
  __syncthreads();

  // 64 column tiles of 16, K = 256: preload all 8 B-fragments, then WMMA chain
  for (int nt = wave; nt < 64; nt += 8) {
    v16h bf[8];
#pragma unroll
    for (int kk = 0; kk < 8; ++kk) bf[kk] = loadB(WinT, 256, nt * 16, kk * 32, lane);
    v8f acc = {};
#pragma unroll
    for (int kk = 0; kk < 8; ++kk) acc = WMMA(loadA(hf, 264, kk * 32, lane), bf[kk], acc);
    // tile-uniform destination split (no per-lane divergence)
    float* dstp = (nt < 32) ? xs_pre : z;
    int col = ((nt < 32) ? nt : nt - 32) * 16 + (lane & 15);
#pragma unroll
    for (int r = 0; r < 8; ++r) {
      int row = l0 + r + ((lane >> 4) << 3);
      dstp[((size_t)b * L_ + row) * DIN + col] = acc[r];
    }
  }
}

// ---------------- 2: causal depthwise conv (k=4) + SiLU, 4 outputs/thread ----------------
__global__ __launch_bounds__(256) void k_conv_silu(const float* xs_pre, const float* cw,
                                                   const float* cb, float* xs, _Float16* xs_f16) {
  int g = blockIdx.x * 256 + threadIdx.x;       // g over B * (L/4) * DIN
  int d = g & 511, lq = (g >> 9) & 255, b = g >> 17;
  int lbase = lq * 4;
  float w0 = cw[d * 4 + 0], w1 = cw[d * 4 + 1], w2 = cw[d * 4 + 2], w3 = cw[d * 4 + 3];
  float bias = cb[d];
  float in[7];
#pragma unroll
  for (int m = 0; m < 7; ++m) {
    int ll = lbase - 3 + m;
    in[m] = (ll >= 0) ? xs_pre[((size_t)b * L_ + ll) * DIN + d] : 0.f;
  }
#pragma unroll
  for (int i = 0; i < 4; ++i) {
    float acc = bias + in[i] * w0 + in[i + 1] * w1 + in[i + 2] * w2 + in[i + 3] * w3;
    float v = acc * sigmoidf_(acc);
    size_t idx = ((size_t)b * L_ + lbase + i) * DIN + d;
    xs[idx] = v;
    xs_f16[idx] = (_Float16)v;
  }
}

// ---------------- 3: x-proj GEMM (512->48) + dt GEMM (16->512) + softplus ----------------
// A tile staged via gfx1250 async global->LDS DMA (ASYNCcnt path)
__global__ __launch_bounds__(256) void k_xproj(const _Float16* xs_f16, const _Float16* WxT,
                                               const _Float16* WdtT, const float* b_dt,
                                               float* dt, float* Bm, float* Cm) {
  __shared__ __align__(16) _Float16 af[16 * 520];
  __shared__ __align__(16) _Float16 dtrA[16 * 40];
  int tid = threadIdx.x, lane = tid & 31, wave = tid >> 5;
  int b = blockIdx.x >> 6, l0 = (blockIdx.x & 63) * 16;

  // async-copy 16x512 f16 tile into padded LDS rows: 1024 chunks of 16B
  {
    const _Float16* src = xs_f16 + ((size_t)b * L_ + l0) * DIN;
#pragma unroll
    for (int it = 0; it < 4; ++it) {
      int q = it * 256 + tid;
      int j = q >> 6, kc = (q & 63) << 3;
      unsigned lds_dst = (unsigned)(size_t)(af + j * 520 + kc);
      unsigned long long ga = (unsigned long long)(size_t)(src + j * 512 + kc);
      async_g2l_b128(lds_dst, ga);
    }
  }
  for (int i = tid; i < 16 * 40; i += 256) dtrA[i] = (_Float16)0.0f;
  wait_asynccnt0();
  __syncthreads();

  // phase 1: xdbl = xs @ W_x (3 column tiles of 16), waves 0..2; 8-deep B preload
  if (wave < 3) {
    v8f acc = {};
#pragma unroll
    for (int kb = 0; kb < 16; kb += 8) {
      v16h bf[8];
#pragma unroll
      for (int q = 0; q < 8; ++q) bf[q] = loadB(WxT, 512, wave * 16, (kb + q) * 32, lane);
#pragma unroll
      for (int q = 0; q < 8; ++q) acc = WMMA(loadA(af, 520, (kb + q) * 32, lane), bf[q], acc);
    }
    int s = lane & 15;
#pragma unroll
    for (int r = 0; r < 8; ++r) {
      int j = r + ((lane >> 4) << 3);
      size_t tok = (size_t)b * L_ + l0 + j;
      float v = acc[r];
      if (wave == 0)      dtrA[j * 40 + s] = (_Float16)v;   // dtr: A-frag staging (K16, zero-pad)
      else if (wave == 1) Bm[tok * DST + s] = v;
      else                Cm[tok * DST + s] = v;
    }
  }
  __syncthreads();

  // phase 2: dt = softplus(dtr @ W_dt + b_dt), 32 column tiles, K=32 (1 WMMA)
  for (int nt = wave; nt < 32; nt += 8) {
    v8f acc = {};
    v16h a = loadA(dtrA, 40, 0, lane);
    v16h bb = loadB(WdtT, 32, nt * 16, 0, lane);
    acc = WMMA(a, bb, acc);
    int col = nt * 16 + (lane & 15);
#pragma unroll
    for (int r = 0; r < 8; ++r) {
      int j = r + ((lane >> 4) << 3);
      dt[((size_t)b * L_ + l0 + j) * DIN + col] = softplusf_(acc[r] + b_dt[col]);
    }
  }
}

// ---------------- 4a: per-chunk scan summaries; Bm chunk staged in LDS ----------------
__global__ __launch_bounds__(256) void k_scan1(const float* dt, const float* xs, const float* Bm,
                                               const float* A_log, float* Aprod, float* Hend) {
  __shared__ float Bs[CL * DST];                // 8KB; block covers one (b,chunk)
  int g = blockIdx.x * 256 + threadIdx.x;
  int d = g & 511, ch = (g >> 9) & 7, b = g >> 12;
  size_t base = (size_t)b * L_ + (size_t)ch * CL;
  for (int i = threadIdx.x; i < CL * DST; i += 256) Bs[i] = Bm[base * DST + i];
  __syncthreads();
  float A[DST], h[DST], Ap[DST];
#pragma unroll
  for (int s = 0; s < DST; ++s) { A[s] = -__expf(A_log[d * DST + s]); h[s] = 0.f; Ap[s] = 1.f; }
  for (int l = 0; l < CL; ++l) {
    size_t t = base + l;
    float dtv = dt[t * DIN + d];
    float dx = dtv * xs[t * DIN + d];
    const float* Bp = Bs + l * DST;
#pragma unroll
    for (int s = 0; s < DST; ++s) {
      float a = __expf(dtv * A[s]);
      h[s] = a * h[s] + dx * Bp[s];
      Ap[s] *= a;
    }
  }
  size_t idx = (((size_t)b * NCH + ch) * DIN + d) * DST;
#pragma unroll
  for (int s = 0; s < DST; ++s) { Aprod[idx + s] = Ap[s]; Hend[idx + s] = h[s]; }
}

// ---------------- 4b: tiny prefix over the 8 chunks ----------------
__global__ __launch_bounds__(256) void k_scan2(const float* Aprod, const float* Hend, float* Hin) {
  int g = blockIdx.x * 256 + threadIdx.x;
  int ds = g & 8191, b = g >> 13;
  float h = 0.f;
#pragma unroll
  for (int ch = 0; ch < NCH; ++ch) {
    size_t i = ((size_t)b * NCH + ch) * (DIN * DST) + ds;
    Hin[i] = h;
    h = Aprod[i] * h + Hend[i];
  }
}

// ---------------- 4c: final scan with correct init state; Bm+Cm staged in LDS ----------------
__global__ __launch_bounds__(256) void k_scan3(const float* dt, const float* xs, const float* Bm,
                                               const float* Cm, const float* A_log,
                                               const float* Hin, float* ys) {
  __shared__ float Bs[CL * DST];
  __shared__ float Cs[CL * DST];
  int g = blockIdx.x * 256 + threadIdx.x;
  int d = g & 511, ch = (g >> 9) & 7, b = g >> 12;
  size_t base = (size_t)b * L_ + (size_t)ch * CL;
  for (int i = threadIdx.x; i < CL * DST; i += 256) {
    Bs[i] = Bm[base * DST + i];
    Cs[i] = Cm[base * DST + i];
  }
  __syncthreads();
  float A[DST], h[DST];
  size_t hidx = (((size_t)b * NCH + ch) * DIN + d) * DST;
#pragma unroll
  for (int s = 0; s < DST; ++s) { A[s] = -__expf(A_log[d * DST + s]); h[s] = Hin[hidx + s]; }
  for (int l = 0; l < CL; ++l) {
    size_t t = base + l;
    float dtv = dt[t * DIN + d];
    float dx = dtv * xs[t * DIN + d];
    const float* Bp = Bs + l * DST;
    const float* Cp = Cs + l * DST;
    float y = 0.f;
#pragma unroll
    for (int s = 0; s < DST; ++s) {
      float a = __expf(dtv * A[s]);
      h[s] = a * h[s] + dx * Bp[s];
      y += h[s] * Cp[s];
    }
    ys[t * DIN + d] = y;
  }
}

// ---------------- 5: gate + out-projection GEMM (512 -> 256), transposed store ----------------
__global__ __launch_bounds__(256) void k_out(const float* ys, const float* xs, const float* z,
                                             const float* Dp, const _Float16* WoutT, float* out) {
  __shared__ __align__(16) _Float16 yf[16 * 520];
  __shared__ __align__(16) float ot[256 * 16];
  int tid = threadIdx.x, lane = tid & 31, wave = tid >> 5;
  int b = blockIdx.x >> 6, l0 = (blockIdx.x & 63) * 16;

  for (int i = tid; i < 16 * 512; i += 256) {
    int j = i >> 9, kd = i & 511;
    size_t t = (size_t)b * L_ + l0 + j;
    float yv = ys[t * DIN + kd] + xs[t * DIN + kd] * Dp[kd];
    float zv = z[t * DIN + kd];
    yf[j * 520 + kd] = (_Float16)(yv * zv * sigmoidf_(zv));
  }
  __syncthreads();

  for (int nt = wave; nt < 16; nt += 8) {
    v8f acc = {};
#pragma unroll
    for (int kb = 0; kb < 16; kb += 8) {
      v16h bf[8];
#pragma unroll
      for (int q = 0; q < 8; ++q) bf[q] = loadB(WoutT, 512, nt * 16, (kb + q) * 32, lane);
#pragma unroll
      for (int q = 0; q < 8; ++q) acc = WMMA(loadA(yf, 520, (kb + q) * 32, lane), bf[q], acc);
    }
    int c = nt * 16 + (lane & 15);
#pragma unroll
    for (int r = 0; r < 8; ++r) {
      int j = r + ((lane >> 4) << 3);
      ot[c * 16 + j] = acc[r];
    }
  }
  __syncthreads();
  // coalesced store: out[b, c, l0+j]
  for (int i = tid; i < 256 * 16; i += 256) {
    int c = i >> 4, j = i & 15;
    out[(size_t)b * C_ * L_ + (size_t)c * L_ + l0 + j] = ot[i];
  }
}

extern "C" void kernel_launch(void* const* d_in, const int* in_sizes, int n_in,
                              void* d_out, int out_size, void* d_ws, size_t ws_size,
                              hipStream_t stream) {
  const float* x      = (const float*)d_in[0];
  const float* ln_w   = (const float*)d_in[1];
  const float* ln_b   = (const float*)d_in[2];
  const float* W_in   = (const float*)d_in[3];
  const float* conv_w = (const float*)d_in[4];
  const float* conv_b = (const float*)d_in[5];
  const float* W_x    = (const float*)d_in[6];
  const float* W_dt   = (const float*)d_in[7];
  const float* b_dt   = (const float*)d_in[8];
  const float* A_log  = (const float*)d_in[9];
  const float* Dp     = (const float*)d_in[10];
  const float* W_out  = (const float*)d_in[11];
  float* out = (float*)d_out;

  char* ws = (char*)d_ws;
  _Float16* WinT  = (_Float16*)(ws + OFF_WINT);
  _Float16* WxT   = (_Float16*)(ws + OFF_WXT);
  _Float16* WdtT  = (_Float16*)(ws + OFF_WDTT);
  _Float16* WoutT = (_Float16*)(ws + OFF_WOUTT);
  float* xs_pre = (float*)(ws + OFF_XSPRE);
  float* z      = (float*)(ws + OFF_Z);
  float* xs     = (float*)(ws + OFF_XS);
  float* dt     = (float*)(ws + OFF_DT);
  float* ys     = (float*)(ws + OFF_YS);
  float* Bm     = (float*)(ws + OFF_BM);
  float* Cm     = (float*)(ws + OFF_CM);
  float* Aprod  = (float*)(ws + OFF_APROD);
  float* Hend   = (float*)(ws + OFF_HEND);
  float* Hin    = (float*)(ws + OFF_HIN);
  _Float16* xs_f16 = (_Float16*)(ws + OFF_XSF16);

  k_prep<<<128, 256, 0, stream>>>(W_in, W_x, W_dt, W_out, WinT, WxT, WdtT, WoutT);
  k_ln_inproj<<<NTTILE, 256, 0, stream>>>(x, ln_w, ln_b, WinT, xs_pre, z);
  k_conv_silu<<<(B_ * (L_ / 4) * DIN) / 256, 256, 0, stream>>>(xs_pre, conv_w, conv_b, xs, xs_f16);
  k_xproj<<<NTTILE, 256, 0, stream>>>(xs_f16, WxT, WdtT, b_dt, dt, Bm, Cm);
  k_scan1<<<(B_ * NCH * DIN) / 256, 256, 0, stream>>>(dt, xs, Bm, A_log, Aprod, Hend);
  k_scan2<<<(B_ * DIN * DST) / 256, 256, 0, stream>>>(Aprod, Hend, Hin);
  k_scan3<<<(B_ * NCH * DIN) / 256, 256, 0, stream>>>(dt, xs, Bm, Cm, A_log, Hin, ys);
  k_out<<<NTTILE, 256, 0, stream>>>(ys, xs, z, Dp, WoutT, out);
}
